// SCAESuite_6236292513846
// MI455X (gfx1250) — matrix operand order repaired and checked by reading, not compiled, not run on hardware
//
#include <hip/hip_runtime.h>

#define B_BATCH 2048
#define KACT    32
#define DDIM    768
#define DICT_N  24576
#define CONN    64
#define LDS_STR 776   // 768 + 8 halves pad -> 1552B row stride, 16B skew mod 256B banks

typedef __attribute__((ext_vector_type(16))) __bf16 bf16x16;
typedef __attribute__((ext_vector_type(8)))  __bf16 bf16x8;
typedef __attribute__((ext_vector_type(8)))  float  f32x8;

__device__ __forceinline__ __bf16 f2bf(float f) {
    unsigned u = __builtin_bit_cast(unsigned, f);
    unsigned r = u + 0x7FFFu + ((u >> 16) & 1u);   // round-to-nearest-even
    unsigned short h = (unsigned short)(r >> 16);
    return __builtin_bit_cast(__bf16, h);
}

// ---------------- b_dec_contrib = up_encoder_w @ b_dec : [DICT] ----------------
__global__ __launch_bounds__(256) void bdc_kernel(const float* __restrict__ up_enc,
                                                  const float* __restrict__ b_dec,
                                                  float* __restrict__ bdc) {
    __shared__ float bvec[DDIM];
    const int tid = threadIdx.x;
    for (int d = tid; d < DDIM; d += 256) bvec[d] = b_dec[d];
    __syncthreads();
    const int wid = tid >> 5, lane = tid & 31;
    const int row = blockIdx.x * 8 + wid;                 // 8 waves, one row each
    const float* rp = up_enc + (long)row * DDIM;
    float s = 0.f;
    for (int d = lane; d < DDIM; d += 32) s += rp[d] * bvec[d];
    #pragma unroll
    for (int off = 16; off > 0; off >>= 1) s += __shfl_down(s, off, 32);
    if (lane == 0) bdc[row] = s;
}

// ---------------- main per-batch kernel: one block per b ----------------
__global__ __launch_bounds__(128) void sae_virtual_kernel(
    const float* __restrict__ up_vals,      // [B,K]
    const float* __restrict__ up_decoder,   // [D, DICT]
    const float* __restrict__ down_encoder, // [DICT, D]
    const int*   __restrict__ up_indices,   // [B,K]
    const int*   __restrict__ down_indices, // [B,K]
    const int*   __restrict__ connections,  // [DICT, C]
    const float* __restrict__ bdc,          // [DICT]
    float*       __restrict__ out)          // [B,K]
{
    extern __shared__ char smem_raw[];
    float*  upv     = (float*)smem_raw;               // 32
    float*  contrib = upv + 32;                       // 32
    float*  mult    = contrib + 32;                   // 32*32
    int*    upidx   = (int*)(mult + KACT * KACT);     // 32
    int*    dnidx   = upidx + 32;                     // 32
    __bf16* a_lds   = (__bf16*)(dnidx + 32);          // 32 * LDS_STR  (act_down, bf16)
    __bf16* b_lds   = a_lds + KACT * LDS_STR;         // 32 * LDS_STR  (act_up^T,  bf16)

    const int tid  = threadIdx.x;
    const int lane = tid & 31;
    const int wid  = tid >> 5;
    const int b    = blockIdx.x;

    // phase 0: metadata + init
    if (tid < KACT) {
        upidx[tid]   = up_indices[b * KACT + tid];
        dnidx[tid]   = down_indices[b * KACT + tid];
        upv[tid]     = up_vals[b * KACT + tid];
        contrib[tid] = 0.f;
    }
    for (int t = tid; t < KACT * KACT; t += 128) mult[t] = 0.f;
    __syncthreads();

    // phase 1: mult[i][j] = #matching connection slots (matches rare -> cheap atomics)
    for (int s = tid; s < KACT * CONN; s += 128) {
        const int i = s >> 6;          // slot-major: /64
        const int c = s & (CONN - 1);
        const int a = connections[(long)dnidx[i] * CONN + c];
        if (a != -1) {
            #pragma unroll
            for (int j = 0; j < KACT; ++j)
                if (a == upidx[j]) atomicAdd(&mult[i * KACT + j], 1.0f);
        }
    }

    // phase 2: stage act_down rows (coalesced) as bf16
    for (int r = wid; r < KACT; r += 4) {
        const float* src = down_encoder + (long)dnidx[r] * DDIM;
        for (int d = lane; d < DDIM; d += 32)
            a_lds[r * LDS_STR + d] = f2bf(src[d]);
    }
    // phase 3: stage act_up columns (strided gather, unavoidable) as bf16
    for (int j = wid; j < KACT; j += 4) {
        const int uj = upidx[j];
        for (int d = lane; d < DDIM; d += 32)
            b_lds[j * LDS_STR + d] = f2bf(up_decoder[(long)d * DICT_N + uj]);
    }
    __syncthreads();

    // phase 4: vw = act_down (32xD) * act_up^T (Dx32) via bf16 WMMA.
    // 4 waves, one 16x16 tile each: wave -> (mi, nj).
    const int mi = wid >> 1, nj = wid & 1;
    const int lsub = lane & 15, hi = lane >> 4;
    const __bf16* arow = a_lds + (mi * 16 + lsub) * LDS_STR;
    const __bf16* brow = b_lds + (nj * 16 + lsub) * LDS_STR;

    f32x8 acc = {};
    for (int kb = 0; kb < DDIM / 32; ++kb) {
        const int k0 = kb * 32;
        // A 16x32 bf16: lanes 0-15 hold K 0-7 & 16-23; lanes 16-31 hold K 8-15 & 24-31
        bf16x8 a0 = *(const bf16x8*)(arow + k0 + hi * 8);
        bf16x8 a1 = *(const bf16x8*)(arow + k0 + 16 + hi * 8);
        // B 32x16 bf16: lanes 0-15 hold K 0-7 & 16-23 of col n; lanes 16-31 hold K 8-15 & 24-31
        bf16x8 b0 = *(const bf16x8*)(brow + k0 + hi * 8);
        bf16x8 b1 = *(const bf16x8*)(brow + k0 + 16 + hi * 8);
        bf16x16 A, Bm;
        #pragma unroll
        for (int e = 0; e < 8; ++e) {
            A[e] = a0[e];  A[e + 8] = a1[e];
            Bm[e] = b0[e]; Bm[e + 8] = b1[e];
        }
        acc = __builtin_amdgcn_wmma_f32_16x16x32_bf16(
                  false, A, false, Bm, (short)0, acc, false, false);
    }

    // phase 5: contributions[i] += sum_j vw[i,j] * mult[i,j] * up_vals[j]
    // C/D layout: VGPR r -> row mi*16 + hi*8 + r ; col nj*16 + lsub
    const int j = nj * 16 + lsub;
    const float uvj = upv[j];
    #pragma unroll
    for (int r = 0; r < 8; ++r) {
        const int i = mi * 16 + hi * 8 + r;
        const float m = mult[i * KACT + j] * uvj;
        if (m != 0.f) atomicAdd(&contrib[i], acc[r] * m);
    }
    __syncthreads();

    if (tid < KACT)
        out[b * KACT + tid] = contrib[tid] + bdc[upidx[tid]];
}

extern "C" void kernel_launch(void* const* d_in, const int* in_sizes, int n_in,
                              void* d_out, int out_size, void* d_ws, size_t ws_size,
                              hipStream_t stream) {
    const float* up_vals  = (const float*)d_in[0];
    const float* up_dec   = (const float*)d_in[1];
    const float* down_enc = (const float*)d_in[2];
    const float* up_enc   = (const float*)d_in[3];
    const float* b_dec    = (const float*)d_in[4];
    const int*   up_idx   = (const int*)d_in[5];
    const int*   dn_idx   = (const int*)d_in[6];
    const int*   conn     = (const int*)d_in[7];
    float* out = (float*)d_out;
    float* bdc = (float*)d_ws;   // 24576 floats of scratch

    bdc_kernel<<<DICT_N / 8, 256, 0, stream>>>(up_enc, b_dec, bdc);

    const size_t lds_bytes = (32 + 32 + KACT * KACT) * sizeof(float)
                           + 64 * sizeof(int)
                           + (size_t)2 * KACT * LDS_STR * sizeof(__bf16);
    sae_virtual_kernel<<<B_BATCH, 128, lds_bytes, stream>>>(
        up_vals, up_dec, down_enc, up_idx, dn_idx, conn, bdc, out);
}
